// HopfieldEnergy_4466765988116
// MI455X (gfx1250) — matrix operand order repaired and checked by reading, not compiled, not run on hardware
//
#include <hip/hip_runtime.h>
#include <hip/hip_bf16.h>

typedef __attribute__((ext_vector_type(16))) _Float16 v16h;
typedef __attribute__((ext_vector_type(8)))  _Float16 v8h;
typedef __attribute__((ext_vector_type(8)))  float    v8f;
typedef __attribute__((ext_vector_type(4)))  float    v4f;

#define N_ROWS 8192     // rows of x
#define M_ROWS 8192     // rows of a (stored patterns)
#define FDIM   256
#define BETA_A 2.0f

static __device__ __forceinline__ v16h cat8(v8h lo, v8h hi) {
  return __builtin_shufflevector(lo, hi, 0,1,2,3,4,5,6,7,8,9,10,11,12,13,14,15);
}

// ---------------------------------------------------------------- init
__global__ void hopfield_init(float* __restrict__ S, float* __restrict__ maxsq) {
  int i = blockIdx.x * blockDim.x + threadIdx.x;
  if (i < N_ROWS) S[i] = 0.0f;
  if (i == 0) *maxsq = 0.0f;
}

// ------------------------------------------------------- normalize + f16
// One wave (32 lanes) per row. Rows [0,8192) -> x, [8192,16384) -> a.
__global__ __launch_bounds__(256) void hopfield_norm(
    const float* __restrict__ x, const float* __restrict__ a,
    _Float16* __restrict__ x16, _Float16* __restrict__ a16,
    float* __restrict__ xsq_half, float* __restrict__ maxsq) {
  int wave = (blockIdx.x * blockDim.x + threadIdx.x) >> 5;
  int lane = threadIdx.x & 31;
  bool isA = wave >= N_ROWS;
  int row  = isA ? (wave - N_ROWS) : wave;
  const float* src = (isA ? a : x) + (size_t)row * FDIM;

  // each lane owns 8 contiguous floats
  v4f v0 = *(const v4f*)(src + lane * 8);
  v4f v1 = *(const v4f*)(src + lane * 8 + 4);
  float ss = v0.x*v0.x + v0.y*v0.y + v0.z*v0.z + v0.w*v0.w
           + v1.x*v1.x + v1.y*v1.y + v1.z*v1.z + v1.w*v1.w;
  #pragma unroll
  for (int off = 16; off >= 1; off >>= 1) ss += __shfl_xor(ss, off, 32);

  float n   = sqrtf(ss);
  float inv = 1.0f / fmaxf(n, 1e-12f);

  _Float16* dst = (isA ? a16 : x16) + (size_t)row * FDIM;
  v8h o;
  o[0]=(_Float16)(v0.x*inv); o[1]=(_Float16)(v0.y*inv);
  o[2]=(_Float16)(v0.z*inv); o[3]=(_Float16)(v0.w*inv);
  o[4]=(_Float16)(v1.x*inv); o[5]=(_Float16)(v1.y*inv);
  o[6]=(_Float16)(v1.z*inv); o[7]=(_Float16)(v1.w*inv);
  *(v8h*)(dst + lane * 8) = o;

  if (lane == 0) {
    float nn = n * inv;                       // norm of the normalized row
    if (isA) atomicMax((unsigned int*)maxsq, __float_as_uint(nn * nn));
    else     xsq_half[row] = 0.5f * (ss * inv * inv);
  }
}

// -------------------------------------- fused GEMM (WMMA f16) + exp-sum
// grid.x = 512 row tiles (16 rows each), grid.y = 8 column chunks (1024 cols).
// Block = 256 threads = 8 waves; wave w covers cols chunk*1024 + it*128 + w*16.
__global__ __launch_bounds__(256) void hopfield_gemm(
    const _Float16* __restrict__ x16, const _Float16* __restrict__ a16,
    float* __restrict__ S) {
  const int lane    = threadIdx.x & 31;
  const int wave    = threadIdx.x >> 5;
  const int rowBase = blockIdx.x * 16;
  const int h       = lane >> 4;      // half of the wave
  const int lm      = lane & 15;      // A: row-in-tile, B: col-in-tile

  // ---- A fragments (x rows), fixed for the whole block: hoist all 8 K-steps.
  // A 16x32 f16 layout: e<8 -> k = h*8+e ; e>=8 -> k = 16 + h*8 + (e-8)
  const _Float16* xrow = x16 + (size_t)(rowBase + lm) * FDIM;
  v16h afrag[8];
  #pragma unroll
  for (int ks = 0; ks < 8; ++ks) {
    int k0 = ks * 32;
    v8h lo = *(const v8h*)(xrow + k0 + h * 8);
    v8h hi = *(const v8h*)(xrow + k0 + 16 + h * 8);
    afrag[ks] = cat8(lo, hi);
  }

  v8f esum = {};  // running sum of exp(2*s) per C element

  #pragma unroll
  for (int it = 0; it < 8; ++it) {
    const int col = blockIdx.y * 1024 + it * 128 + wave * 16;
    const _Float16* arow = a16 + (size_t)(col + lm) * FDIM;
    if (it + 1 < 8)   // prefetch next B tile row -> global_prefetch_b8
      __builtin_prefetch(a16 + (size_t)(col + 128 + lm) * FDIM, 0, 0);

    v8f c = {};
    #pragma unroll
    for (int ks = 0; ks < 8; ++ks) {
      int k0 = ks * 32;
      // B 32x16 f16 layout: e<8 -> k = h*8+e ; e>=8 -> k = 16 + h*8 + (e-8)
      v8h blo = *(const v8h*)(arow + k0 + h * 8);
      v8h bhi = *(const v8h*)(arow + k0 + 16 + h * 8);
      v16h b = cat8(blo, bhi);
      c = __builtin_amdgcn_wmma_f32_16x16x32_f16(
            /*neg_a=*/false, afrag[ks], /*neg_b=*/false, b,
            /*c_mod=*/(short)0, c, /*reuse_a=*/false, /*reuse_b=*/false);
    }
    // epilogue: s in [-1,1] -> exp(2s) safe without max-subtraction
    #pragma unroll
    for (int v = 0; v < 8; ++v) esum[v] += __expf(BETA_A * c[v]);
  }

  // reduce over the 16 columns (lanes within each half)
  #pragma unroll
  for (int off = 1; off <= 8; off <<= 1) {
    #pragma unroll
    for (int v = 0; v < 8; ++v) esum[v] += __shfl_xor(esum[v], off, 32);
  }
  if (lm == 0) {
    // C layout: VGPR v, lanes 0-15 -> M=v ; lanes 16-31 -> M=v+8
    int row = rowBase + h * 8;
    #pragma unroll
    for (int v = 0; v < 8; ++v) atomicAdd(&S[row + v], esum[v]);
  }
}

// ---------------------------------------------------------------- final
__global__ void hopfield_finalize(const float* __restrict__ S,
                                  const float* __restrict__ xsq_half,
                                  const float* __restrict__ maxsq,
                                  float* __restrict__ out) {
  int i = blockIdx.x * blockDim.x + threadIdx.x;
  if (i < N_ROWS) {
    float lse      = __logf(S[i]);                       // logsumexp(2*s) (no shift needed)
    float a_energy = -(lse - __logf((float)M_ROWS)) / BETA_A;
    out[i] = a_energy + xsq_half[i] + 0.5f * (*maxsq);
  }
}

extern "C" void kernel_launch(void* const* d_in, const int* in_sizes, int n_in,
                              void* d_out, int out_size, void* d_ws, size_t ws_size,
                              hipStream_t stream) {
  const float* x = (const float*)d_in[0];
  const float* a = (const float*)d_in[1];
  float* out = (float*)d_out;

  char* ws = (char*)d_ws;
  const size_t szHalf = (size_t)N_ROWS * FDIM * sizeof(_Float16);  // 4 MB
  _Float16* x16   = (_Float16*)(ws);
  _Float16* a16   = (_Float16*)(ws + szHalf);
  float*    S     = (float*)(ws + 2 * szHalf);
  float*    xsqh  = (float*)(ws + 2 * szHalf + N_ROWS * sizeof(float));
  float*    maxsq = (float*)(ws + 2 * szHalf + 2 * N_ROWS * sizeof(float));

  hopfield_init<<<N_ROWS / 256, 256, 0, stream>>>(S, maxsq);
  hopfield_norm<<<(2 * N_ROWS * 32) / 256, 256, 0, stream>>>(x, a, x16, a16, xsqh, maxsq);
  dim3 grid(N_ROWS / 16, M_ROWS / 1024);
  hopfield_gemm<<<grid, 256, 0, stream>>>(x16, a16, S);
  hopfield_finalize<<<N_ROWS / 256, 256, 0, stream>>>(S, xsqh, maxsq, out);
}
